// DiTBlock_85822036509267
// MI455X (gfx1250) — compile-verified, matrix operand order checked by reading, not run on hardware
//
#include <hip/hip_runtime.h>
#include <stdint.h>

typedef unsigned short u16;
typedef __attribute__((ext_vector_type(8)))  float  v8f;
typedef __attribute__((ext_vector_type(8)))  __bf16 v8bf;
typedef __attribute__((ext_vector_type(16))) __bf16 v16bf;

#define DEVFN static __device__ __forceinline__

// ---------- bf16 helpers (round-to-nearest-even, stored as u16) ----------
DEVFN u16 f2bfu(float f) {
  unsigned u = __builtin_bit_cast(unsigned, f);
  unsigned r = u + 0x7FFFu + ((u >> 16) & 1u);
  return (u16)(r >> 16);
}
DEVFN float bf2f(u16 h) {
  unsigned u = ((unsigned)h) << 16;
  return __builtin_bit_cast(float, u);
}

// ---------- WMMA helpers ----------
DEVFN v8f wmma_zero() { v8f z = {}; return z; }

DEVFN v8f wmma_bf16(v16bf a, v16bf b, v8f c) {
  // D = A(16x32 bf16) * B(32x16 bf16) + C(16x16 f32)
  return __builtin_amdgcn_wmma_f32_16x16x32_bf16(false, a, false, b, (short)0, c,
                                                 false, false);
}

// Load a 16x32 bf16 fragment in the 16-bit A-matrix layout (CDNA5 ISA):
//   lane = row (lane&15); lanes<16 hold K=0..7 and 16..23, lanes>=16 hold K=8..15 and 24..31.
DEVFN v16bf load_frag(const u16* base, int rowStride, int lane) {
  int r  = lane & 15;
  int k1 = (lane & 16) ? 8 : 0;
  const v8bf* p1 = (const v8bf*)(base + r * rowStride + k1);
  const v8bf* p2 = (const v8bf*)(base + r * rowStride + k1 + 16);
  v8bf lo = *p1, hi = *p2;
  return __builtin_shufflevector(lo, hi, 0, 1, 2, 3, 4, 5, 6, 7,
                                         8, 9, 10, 11, 12, 13, 14, 15);
}

// ---------- CDNA5 async global->LDS copy (ASYNCcnt path) ----------
DEVFN void async_ld16(const u16* lds_dst, const u16* gsrc) {
  unsigned loff = (unsigned)(uintptr_t)lds_dst;              // LDS byte offset
  unsigned long long g = (unsigned long long)(uintptr_t)gsrc;
  asm volatile("global_load_async_to_lds_b128 %0, %1, off"
               :: "v"(loff), "v"(g) : "memory");
}
DEVFN void wait_async0() { asm volatile("s_wait_asynccnt 0" ::: "memory"); }
DEVFN void wait_async4() { asm volatile("s_wait_asynccnt 4" ::: "memory"); }

// ---------- block reduce ----------
DEVFN float block_sum(float v, float* sbuf) {
  int t = threadIdx.x;
  sbuf[t] = v;
  __syncthreads();
#pragma unroll
  for (int s = 128; s > 0; s >>= 1) {
    if (t < s) sbuf[t] += sbuf[t + s];
    __syncthreads();
  }
  float r = sbuf[0];
  __syncthreads();
  return r;
}

// =====================================================================
// 1) NormalizedDense weight prep
// =====================================================================
__global__ __launch_bounds__(256) void k_rownorm_t(const float* __restrict__ w,
                                                   u16* __restrict__ wt, int R, int C) {
  __shared__ float sbuf[256];
  int r = blockIdx.x, t = threadIdx.x;
  const float* row = w + (size_t)r * C;
  float acc = 0.f;
  for (int c = t; c < C; c += 256) { float v = row[c]; acc += v * v; }
  float nrm2 = block_sum(acc, sbuf);
  float scale = rsqrtf((float)R) / (sqrtf(nrm2) * sqrtf((float)C) + 1e-4f);
  for (int c = t; c < C; c += 256)
    wt[(size_t)c * R + r] = f2bfu(row[c] * scale);
}

// =====================================================================
// 2) conditioning: cc[b][j] = sum_i mp_silu(c[b][i]) * wn_cond[i][j]
// =====================================================================
__global__ __launch_bounds__(256) void k_cond(const float* __restrict__ c,
                                              const u16* __restrict__ wtc,  // [2048][1024]
                                              float* __restrict__ cc) {
  int gid = blockIdx.x * 256 + threadIdx.x;  // 0..8191
  int b = gid >> 11, j = gid & 2047;
  const float* cb = c + b * 1024;
  const u16* wr = wtc + (size_t)j * 1024;
  float acc = 0.f;
  for (int i = 0; i < 1024; ++i) {
    float x = cb[i];
    float s = (x / (1.f + __expf(-x))) * (1.f / 0.596f);
    acc += s * bf2f(wr[i]);
  }
  cc[b * 2048 + j] = acc;
}

// =====================================================================
// 3) x_cond = bf16( pixel_norm(x) * (1+gain) + shift )   (row = token)
// =====================================================================
__global__ __launch_bounds__(256) void k_xcond(const float* __restrict__ xin,
                                               const float* __restrict__ cc,
                                               u16* __restrict__ xc) {
  __shared__ float sbuf[256];
  int row = blockIdx.x, t = threadIdx.x, b = row >> 10;
  const float* xr = xin + (size_t)row * 1024;
  float v[4]; float acc = 0.f;
#pragma unroll
  for (int i = 0; i < 4; i++) { v[i] = xr[t + 256 * i]; acc += v[i] * v[i]; }
  float pn = rsqrtf(block_sum(acc, sbuf) * (1.f / 1024.f) + 1e-4f);
  const float* gp = cc + b * 2048;
#pragma unroll
  for (int i = 0; i < 4; i++) {
    int cidx = t + 256 * i;
    xc[(size_t)row * 1024 + cidx] = f2bfu(v[i] * pn * (1.f + gp[cidx]) + gp[1024 + cidx]);
  }
}

// =====================================================================
// 4) Generic bf16 WMMA GEMM:  C[M][N] (f32) = A[M][K] (bf16) * Bt[N][K]^T (bf16)
//    128x128x32 block tile, 8 waves (2x4), wave tile 64x32.
//    Double-buffered LDS staging via global_load_async_to_lds_b128, with the
//    last k-step peeled so the steady-state loop is branch-free.
// =====================================================================
__global__ __launch_bounds__(256) void k_gemm(const u16* __restrict__ A,
                                              const u16* __restrict__ Bt,
                                              float* __restrict__ C,
                                              int M, int N, int K) {
  __shared__ alignas(16) u16 sA[2][128 * 40];
  __shared__ alignas(16) u16 sB[2][128 * 40];
  int t = threadIdx.x, lane = t & 31, wid = t >> 5;
  int bm = blockIdx.y * 128, bn = blockIdx.x * 128;
  int wr = (wid & 1) * 64, wc = (wid >> 1) * 32;

  // per-thread copy assignment: chunks t and t+256 of 512 (row = c>>2, kofs = (c&3)*8)
  int r0 = t >> 2;                 // 0..63
  int r1 = r0 + 64;                // 64..127
  int kco = (t & 3) * 8;           // 0/8/16/24
  int o0 = r0 * 40 + kco;          // LDS element offsets (same pattern for A and B)
  int o1 = r1 * 40 + kco;
  const u16* gA0 = A + (size_t)(bm + r0) * K + kco;   // hoisted: += 32 per k-step
  const u16* gA1 = A + (size_t)(bm + r1) * K + kco;
  const u16* gB0 = Bt + (size_t)(bn + r0) * K + kco;
  const u16* gB1 = Bt + (size_t)(bn + r1) * K + kco;

  v8f acc[4][2];
#pragma unroll
  for (int i = 0; i < 4; i++)
#pragma unroll
    for (int j = 0; j < 2; j++) acc[i][j] = wmma_zero();

  auto prefetch = [&](int buf, int k0) {
    async_ld16(&sA[buf][o0], gA0 + k0);
    async_ld16(&sA[buf][o1], gA1 + k0);
    async_ld16(&sB[buf][o0], gB0 + k0);
    async_ld16(&sB[buf][o1], gB1 + k0);
  };
  auto compute = [&](int cur) {
    v16bf af[4], bfr[2];
#pragma unroll
    for (int i = 0; i < 4; i++) af[i] = load_frag(&sA[cur][(wr + i * 16) * 40], 40, lane);
#pragma unroll
    for (int j = 0; j < 2; j++) bfr[j] = load_frag(&sB[cur][(wc + j * 16) * 40], 40, lane);
#pragma unroll
    for (int i = 0; i < 4; i++)
#pragma unroll
      for (int j = 0; j < 2; j++) acc[i][j] = wmma_bf16(af[i], bfr[j], acc[i][j]);
  };

  const int KT = K >> 5;           // K is a multiple of 32, KT >= 2
  prefetch(0, 0);
  for (int kt = 0; kt < KT - 1; ++kt) {   // branch-free steady state
    int cur = kt & 1;
    prefetch(cur ^ 1, (kt + 1) * 32);
    wait_async4();   // 4 newest (next tile) stay in flight; current tile landed
    __syncthreads();
    compute(cur);
    __syncthreads();
  }
  wait_async0();                   // peeled last k-step
  __syncthreads();
  compute((KT - 1) & 1);

  int rbase = bm + wr + ((lane & 16) ? 8 : 0);
  int cbase = bn + wc + (lane & 15);
#pragma unroll
  for (int i = 0; i < 4; i++)
#pragma unroll
    for (int j = 0; j < 2; j++)
#pragma unroll
      for (int e = 0; e < 8; e++)
        C[(size_t)(rbase + i * 16 + e) * N + cbase + j * 16] = acc[i][j][e];
}

// =====================================================================
// 5) qkv post-norm: pixel_norm over each 1024-chunk, then per-head L2 norm for q,k.
//    q,k -> [B*NH][N][DH] bf16 ; v -> transposed [B*NH][DH][N] bf16.
// =====================================================================
__global__ __launch_bounds__(256) void k_qkvnorm(const float* __restrict__ qkv,  // [T][3072]
                                                 u16* __restrict__ q, u16* __restrict__ k,
                                                 u16* __restrict__ vt) {
  __shared__ float sbuf[256];
  __shared__ float hs[32];  // [0..15]=q heads, [16..31]=k heads
  int row = blockIdx.x, t = threadIdx.x;
  int b = row >> 10, n = row & 1023;
  const float* base = qkv + (size_t)row * 3072;
  float lq = 0, lk = 0, lv = 0;
  float qv[4], kv[4], vv[4];
#pragma unroll
  for (int i = 0; i < 4; i++) {
    int idx = t * 4 + i;
    qv[i] = base[idx];
    kv[i] = base[1024 + idx];
    vv[i] = base[2048 + idx];
    lq += qv[i] * qv[i]; lk += kv[i] * kv[i]; lv += vv[i] * vv[i];
  }
  float pq = rsqrtf(block_sum(lq, sbuf) * (1.f / 1024.f) + 1e-4f);
  float pk = rsqrtf(block_sum(lk, sbuf) * (1.f / 1024.f) + 1e-4f);
  float pv = rsqrtf(block_sum(lv, sbuf) * (1.f / 1024.f) + 1e-4f);
  if (t < 32) hs[t] = 0.f;
  __syncthreads();
  int head = (t * 4) >> 6;
  float aq = 0, ak = 0;
#pragma unroll
  for (int i = 0; i < 4; i++) { float a = qv[i] * pq, c2 = kv[i] * pk; aq += a * a; ak += c2 * c2; }
  atomicAdd(&hs[head], aq);
  atomicAdd(&hs[16 + head], ak);
  __syncthreads();
  float rq = rsqrtf(hs[head] + 1e-6f);
  float rk = rsqrtf(hs[16 + head] + 1e-6f);
#pragma unroll
  for (int i = 0; i < 4; i++) {
    int idx = t * 4 + i; int h = idx >> 6, d = idx & 63;
    size_t bh = (size_t)(b * 16 + h);
    q[bh * 65536 + (size_t)n * 64 + d]    = f2bfu(qv[i] * pq * rq);
    k[bh * 65536 + (size_t)n * 64 + d]    = f2bfu(kv[i] * pk * rk);
    vt[bh * 65536 + (size_t)d * 1024 + n] = f2bfu(vv[i] * pv);
  }
}

// =====================================================================
// 6) flash attention: per wave 16 queries x full 1024 keys, WMMA for QK^T and PV.
// =====================================================================
__global__ __launch_bounds__(256) void k_attn(const u16* __restrict__ q,
                                              const u16* __restrict__ kk,
                                              const u16* __restrict__ vt,
                                              u16* __restrict__ ctx_out) {
  __shared__ alignas(16) u16 Pb[8][16 * 40];
  int t = threadIdx.x, lane = t & 31, wid = t >> 5;
  int bh = blockIdx.x >> 3, qt = blockIdx.x & 7;
  int b = bh >> 4, h = bh & 15;
  int q0 = qt * 128 + wid * 16;
  const u16* qb = q + (size_t)bh * 65536 + (size_t)q0 * 64;
  const u16* kb = kk + (size_t)bh * 65536;
  const u16* vb = vt + (size_t)bh * 65536;
  v16bf aq0 = load_frag(qb, 64, lane);       // dims 0..31
  v16bf aq1 = load_frag(qb + 32, 64, lane);  // dims 32..63
  v8f ctx[4];
#pragma unroll
  for (int f = 0; f < 4; f++) ctx[f] = wmma_zero();
  float m[8], l[8];
#pragma unroll
  for (int j = 0; j < 8; j++) { m[j] = -1e30f; l[j] = 0.f; }
  u16* pb = &Pb[wid][0];
  int prow = (lane & 16) ? 8 : 0;
  int pcol = lane & 15;

  for (int kc = 0; kc < 1024; kc += 32) {
    v16bf bk;
    v8f s0 = wmma_zero(), s1 = wmma_zero();
    bk = load_frag(kb + (size_t)kc * 64, 64, lane);             s0 = wmma_bf16(aq0, bk, s0);
    bk = load_frag(kb + (size_t)kc * 64 + 32, 64, lane);        s0 = wmma_bf16(aq1, bk, s0);
    bk = load_frag(kb + (size_t)(kc + 16) * 64, 64, lane);      s1 = wmma_bf16(aq0, bk, s1);
    bk = load_frag(kb + (size_t)(kc + 16) * 64 + 32, 64, lane); s1 = wmma_bf16(aq1, bk, s1);

#pragma unroll
    for (int j = 0; j < 8; j++) {
      float a0 = s0[j] * 0.125f, a1 = s1[j] * 0.125f;  // 1/sqrt(DH)
      float mx = fmaxf(a0, a1);
#pragma unroll
      for (int o = 8; o > 0; o >>= 1) mx = fmaxf(mx, __shfl_xor(mx, o, 32));
      float mn = fmaxf(m[j], mx);
      float fac = __expf(m[j] - mn);
      float p0 = __expf(a0 - mn), p1 = __expf(a1 - mn);
      float rs = p0 + p1;
#pragma unroll
      for (int o = 8; o > 0; o >>= 1) rs += __shfl_xor(rs, o, 32);
      l[j] = l[j] * fac + rs;
      m[j] = mn;
#pragma unroll
      for (int f = 0; f < 4; f++) ctx[f][j] *= fac;
      pb[(prow + j) * 40 + pcol]      = f2bfu(p0);
      pb[(prow + j) * 40 + 16 + pcol] = f2bfu(p1);
    }
    asm volatile("" ::: "memory");
    __builtin_amdgcn_wave_barrier();  // wave-private LDS: order P stores vs A-frag reads
    v16bf ap = load_frag(pb, 40, lane);
#pragma unroll
    for (int f = 0; f < 4; f++) {
      v16bf bv = load_frag(vb + (size_t)(f * 16) * 1024 + kc, 1024, lane);
      ctx[f] = wmma_bf16(ap, bv, ctx[f]);
    }
  }

  float inv[8];
#pragma unroll
  for (int j = 0; j < 8; j++) inv[j] = 1.f / l[j];
  int rbase = b * 1024 + q0 + prow;
  int cb = h * 64 + pcol;
#pragma unroll
  for (int f = 0; f < 4; f++)
#pragma unroll
    for (int j = 0; j < 8; j++)
      ctx_out[(size_t)(rbase + j) * 1024 + cb + f * 16] = f2bfu(ctx[f][j] * inv[j]);
}

// =====================================================================
// 7) residual 1
// =====================================================================
__global__ __launch_bounds__(256) void k_resid1(const float* __restrict__ xin,
                                                const float* __restrict__ y,
                                                const float* __restrict__ cc,
                                                const float* __restrict__ gain_s,
                                                float* __restrict__ x1,
                                                u16* __restrict__ xc2) {
  __shared__ float sbuf[256];
  int row = blockIdx.x, t = threadIdx.x, b = row >> 10;
  const float* yr = y + (size_t)row * 1024;
  const float* xr = xin + (size_t)row * 1024;
  float yv[4]; float acc = 0.f;
#pragma unroll
  for (int i = 0; i < 4; i++) { yv[i] = yr[t + 256 * i]; acc += yv[i] * yv[i]; }
  float pny = rsqrtf(block_sum(acc, sbuf) * (1.f / 1024.f) + 1e-4f);
  float ga = __expf(gain_s[0]);
  const float inv_denom = 1.f / sqrtf(0.58f);  // t=0.3
  float x1v[4]; float acc2 = 0.f;
#pragma unroll
  for (int i = 0; i < 4; i++) {
    float xa = yv[i] * pny * ga;
    float nv = (0.7f * xr[t + 256 * i] + 0.3f * xa) * inv_denom;
    x1v[i] = nv; acc2 += nv * nv;
    x1[(size_t)row * 1024 + t + 256 * i] = nv;
  }
  float pn = rsqrtf(block_sum(acc2, sbuf) * (1.f / 1024.f) + 1e-4f);
  const float* gp = cc + b * 2048;
#pragma unroll
  for (int i = 0; i < 4; i++) {
    int cidx = t + 256 * i;
    xc2[(size_t)row * 1024 + cidx] = f2bfu(x1v[i] * pn * (1.f + gp[cidx]) + gp[1024 + cidx]);
  }
}

// =====================================================================
// 8) h = bf16( mp_silu(pixel_norm(h_raw)) ), row width 4096
// =====================================================================
__global__ __launch_bounds__(256) void k_hact(const float* __restrict__ hraw,
                                              u16* __restrict__ h) {
  __shared__ float sbuf[256];
  int row = blockIdx.x, t = threadIdx.x;
  const float* hr = hraw + (size_t)row * 4096;
  float v[16]; float acc = 0.f;
#pragma unroll
  for (int i = 0; i < 16; i++) { v[i] = hr[t + 256 * i]; acc += v[i] * v[i]; }
  float pn = rsqrtf(block_sum(acc, sbuf) * (1.f / 4096.f) + 1e-4f);
#pragma unroll
  for (int i = 0; i < 16; i++) {
    float z = v[i] * pn;
    float s = (z / (1.f + __expf(-z))) * (1.f / 0.596f);
    h[(size_t)row * 4096 + t + 256 * i] = f2bfu(s);
  }
}

// =====================================================================
// 9) residual 2
// =====================================================================
__global__ __launch_bounds__(256) void k_resid2(const float* __restrict__ x1,
                                                const float* __restrict__ y,
                                                const float* __restrict__ gain_s,
                                                float* __restrict__ out) {
  __shared__ float sbuf[256];
  int row = blockIdx.x, t = threadIdx.x;
  const float* yr = y + (size_t)row * 1024;
  const float* xr = x1 + (size_t)row * 1024;
  float yv[4]; float acc = 0.f;
#pragma unroll
  for (int i = 0; i < 4; i++) { yv[i] = yr[t + 256 * i]; acc += yv[i] * yv[i]; }
  float pn = rsqrtf(block_sum(acc, sbuf) * (1.f / 1024.f) + 1e-4f);
  float gm = __expf(gain_s[0]);
  const float inv_denom = 1.f / sqrtf(0.58f);
#pragma unroll
  for (int i = 0; i < 4; i++) {
    float xm = yv[i] * pn * gm;
    out[(size_t)row * 1024 + t + 256 * i] = (0.7f * xr[t + 256 * i] + 0.3f * xm) * inv_denom;
  }
}

// =====================================================================
extern "C" void kernel_launch(void* const* d_in, const int* in_sizes, int n_in,
                              void* d_out, int out_size, void* d_ws, size_t ws_size,
                              hipStream_t stream) {
  (void)in_sizes; (void)n_in; (void)out_size; (void)ws_size;
  const float* x       = (const float*)d_in[0];
  const float* c       = (const float*)d_in[1];
  const float* w_cond  = (const float*)d_in[2];
  const float* w_qkv   = (const float*)d_in[3];
  const float* w_out   = (const float*)d_in[4];
  const float* w_mlp1  = (const float*)d_in[5];
  const float* w_mlp2  = (const float*)d_in[6];
  const float* a_gain  = (const float*)d_in[7];
  const float* m_gain  = (const float*)d_in[8];
  float* out = (float*)d_out;

  char* ws = (char*)d_ws;
  size_t off = 0;
  auto alloc = [&](size_t bytes) { size_t r = off; off += (bytes + 255) & ~(size_t)255; return r; };
  const size_t T = 4096;  // B*N tokens

  u16* wt_qkv  = (u16*)(ws + alloc(3072ull * 1024 * 2));
  u16* wt_out  = (u16*)(ws + alloc(1024ull * 1024 * 2));
  u16* wt_mlp1 = (u16*)(ws + alloc(4096ull * 1024 * 2));
  u16* wt_mlp2 = (u16*)(ws + alloc(1024ull * 4096 * 2));
  u16* wt_cond = (u16*)(ws + alloc(2048ull * 1024 * 2));
  float* cc    = (float*)(ws + alloc(4 * 2048 * 4));
  u16* xc      = (u16*)(ws + alloc(T * 1024 * 2));       // x_cond / xc2 (reused)
  float* big   = (float*)(ws + alloc(T * 4096 * 4));     // qkv raw, later h_raw
  u16* qbuf    = (u16*)(ws + alloc(T * 4096 * 2));       // q|k|vT, later h bf16
  u16* kbuf    = qbuf + T * 1024;
  u16* vtbuf   = qbuf + 2 * T * 1024;
  u16* hbuf    = qbuf;                                   // reuse after attention
  u16* ctx     = (u16*)(ws + alloc(T * 1024 * 2));
  float* ybuf  = (float*)(ws + alloc(T * 1024 * 4));     // proj out / mlp out (reused)
  float* x1    = (float*)(ws + alloc(T * 1024 * 4));

  // weight normalization (bf16, transposed)
  k_rownorm_t<<<1024, 256, 0, stream>>>(w_qkv,  wt_qkv,  1024, 3072);
  k_rownorm_t<<<1024, 256, 0, stream>>>(w_out,  wt_out,  1024, 1024);
  k_rownorm_t<<<1024, 256, 0, stream>>>(w_mlp1, wt_mlp1, 1024, 4096);
  k_rownorm_t<<<4096, 256, 0, stream>>>(w_mlp2, wt_mlp2, 4096, 1024);
  k_rownorm_t<<<1024, 256, 0, stream>>>(w_cond, wt_cond, 1024, 2048);

  // conditioning -> gain/shift
  k_cond<<<32, 256, 0, stream>>>(c, wt_cond, cc);

  // attention branch
  k_xcond<<<4096, 256, 0, stream>>>(x, cc, xc);
  k_gemm<<<dim3(24, 32), 256, 0, stream>>>(xc, wt_qkv, big, 4096, 3072, 1024);
  k_qkvnorm<<<4096, 256, 0, stream>>>(big, qbuf, kbuf, vtbuf);
  k_attn<<<512, 256, 0, stream>>>(qbuf, kbuf, vtbuf, ctx);
  k_gemm<<<dim3(8, 32), 256, 0, stream>>>(ctx, wt_out, ybuf, 4096, 1024, 1024);
  k_resid1<<<4096, 256, 0, stream>>>(x, ybuf, cc, a_gain, x1, xc);

  // MLP branch
  k_gemm<<<dim3(32, 32), 256, 0, stream>>>(xc, wt_mlp1, big, 4096, 4096, 1024);
  k_hact<<<4096, 256, 0, stream>>>(big, hbuf);
  k_gemm<<<dim3(8, 32), 256, 0, stream>>>(hbuf, wt_mlp2, ybuf, 4096, 1024, 4096);
  k_resid2<<<4096, 256, 0, stream>>>(x1, ybuf, m_gain, out);
}
